// RNN_39951785787977
// MI455X (gfx1250) — matrix operand order repaired and checked by reading, not compiled, not run on hardware
//
#include <hip/hip_runtime.h>
#include <hip/hip_bf16.h>
#include <math.h>

#define EMB    1536
#define HID    1536
#define VOCAB  10000
#define T_STEPS 128
#define BATCH  128
#define LAYERS 2

typedef __attribute__((ext_vector_type(16))) __bf16 v16bf;
typedef __attribute__((ext_vector_type(8)))  __bf16 v8bf;
typedef __attribute__((ext_vector_type(8)))  float  v8f;
typedef __attribute__((ext_vector_type(4)))  float  v4f;

// ---------------------------------------------------------------------------
// WMMA fragment loaders (wave32, v_wmma_f32_16x16x32_bf16 layouts)
// A 16x32 bf16: lane m = lane&15; kb = (lane>>4)*8; elems 0..7 = K[kb..kb+7],
//               elems 8..15 = K[kb+16..kb+23]  (row-major A, stride ldk)
// B 32x16 bf16: lane n = lane&15; kb = (lane>>4)*8; elems 0..7 = K[kb..kb+7],
//               elems 8..15 = K[kb+16..kb+23]
//               (loaded from pre-transposed B^T, row-major N x K)
// C/D 16x16 f32: lane n = lane&15; vgpr r holds row m = r + 8*(lane>>4)
// ---------------------------------------------------------------------------

__device__ __forceinline__ v16bf load_a_frag(const __bf16* __restrict__ A,
                                             int m0, int k0, int ldk) {
    const int lane = threadIdx.x & 31;
    const __bf16* p = A + (size_t)(m0 + (lane & 15)) * ldk + k0 + ((lane >> 4) << 3);
    v8bf lo = *(const v8bf*)(p);
    v8bf hi = *(const v8bf*)(p + 16);
    v16bf r;
#pragma unroll
    for (int i = 0; i < 8; ++i) { r[i] = lo[i]; r[i + 8] = hi[i]; }
    return r;
}

__device__ __forceinline__ v16bf load_b_frag(const __bf16* __restrict__ BT,
                                             int n0, int k0, int ldk) {
    const int lane = threadIdx.x & 31;
    const __bf16* p = BT + (size_t)(n0 + (lane & 15)) * ldk + k0 + ((lane >> 4) << 3);
    v8bf lo = *(const v8bf*)(p);
    v8bf hi = *(const v8bf*)(p + 16);
    v16bf r;
#pragma unroll
    for (int i = 0; i < 8; ++i) { r[i] = lo[i]; r[i + 8] = hi[i]; }
    return r;
}

// A-fragment fused with the embedding gather: row -> lut[inputs[row]]*sqrt(EMB),
// converted f32 -> bf16 in registers (avoids materializing 50 MB of emb).
__device__ __forceinline__ v16bf load_a_frag_embed(const float* __restrict__ lut,
                                                   const int* __restrict__ tok,
                                                   int m0, int k0, float scale) {
    const int lane = threadIdx.x & 31;
    const int row  = m0 + (lane & 15);
    const float* base = lut + (size_t)tok[row] * EMB + k0 + ((lane >> 4) << 3);
    v4f a0 = *(const v4f*)(base);
    v4f a1 = *(const v4f*)(base + 4);
    v4f a2 = *(const v4f*)(base + 16);
    v4f a3 = *(const v4f*)(base + 20);
    v16bf r;
#pragma unroll
    for (int i = 0; i < 4; ++i) {
        r[i]      = (__bf16)(a0[i] * scale);
        r[4 + i]  = (__bf16)(a1[i] * scale);
        r[8 + i]  = (__bf16)(a2[i] * scale);
        r[12 + i] = (__bf16)(a3[i] * scale);
    }
    return r;
}

// ---------------------------------------------------------------------------
// Prep kernels
// ---------------------------------------------------------------------------

// WT[n*K + k] = (bf16) W[k*N + n]   (weights -> bf16, transposed to N x K)
__global__ void cvt_transpose_kernel(const float* __restrict__ W,
                                     __bf16* __restrict__ WT, int K, int N) {
    size_t total = (size_t)K * N;
    for (size_t i = (size_t)blockIdx.x * blockDim.x + threadIdx.x; i < total;
         i += (size_t)gridDim.x * blockDim.x) {
        size_t n = i / K, k = i % K;
        WT[i] = (__bf16)W[k * (size_t)N + n];
    }
}

__global__ void init_hidden_kernel(const float* __restrict__ src,
                                   float* __restrict__ hf,
                                   __bf16* __restrict__ hbf, size_t n) {
    for (size_t i = (size_t)blockIdx.x * blockDim.x + threadIdx.x; i < n;
         i += (size_t)gridDim.x * blockDim.x) {
        float v = src[i];
        hf[i]  = v;
        hbf[i] = (__bf16)v;
    }
}

__global__ void cvt_bf16_kernel(const float* __restrict__ src,
                                __bf16* __restrict__ dst, size_t n) {
    for (size_t i = (size_t)blockIdx.x * blockDim.x + threadIdx.x; i < n;
         i += (size_t)gridDim.x * blockDim.x)
        dst[i] = (__bf16)src[i];
}

__global__ void copy_f32_kernel(const float* __restrict__ src,
                                float* __restrict__ dst, size_t n) {
    for (size_t i = (size_t)blockIdx.x * blockDim.x + threadIdx.x; i < n;
         i += (size_t)gridDim.x * blockDim.x)
        dst[i] = src[i];
}

// ---------------------------------------------------------------------------
// xw = (lut[inputs]*sqrt(EMB)) @ Wx     M=16384 N=1536 K=1536, bf16 WMMA
// one wave per 16(M) x 64(N) slab: A gathered once per K-chunk, 4 independent
// accumulator chains hide WMMA latency
// ---------------------------------------------------------------------------
__global__ void embed_xw_gemm_kernel(const float* __restrict__ lut,
                                     const int* __restrict__ inputs,
                                     const __bf16* __restrict__ WxT,
                                     float* __restrict__ xw) {
    const int NG = HID / 64;  // 24 N-groups of 4 tiles
    const int gw = (int)((blockIdx.x * blockDim.x + threadIdx.x) >> 5);
    const int m0 = (gw / NG) * 16;
    const int n0 = (gw % NG) * 64;
    const float scale = 39.1918358845f;  // sqrt(1536)

    v8f acc0 = {}, acc1 = {}, acc2 = {}, acc3 = {};
    for (int k0 = 0; k0 < EMB; k0 += 32) {
        v16bf a  = load_a_frag_embed(lut, inputs, m0, k0, scale);
        v16bf b0 = load_b_frag(WxT, n0,      k0, EMB);
        v16bf b1 = load_b_frag(WxT, n0 + 16, k0, EMB);
        v16bf b2 = load_b_frag(WxT, n0 + 32, k0, EMB);
        v16bf b3 = load_b_frag(WxT, n0 + 48, k0, EMB);
        acc0 = __builtin_amdgcn_wmma_f32_16x16x32_bf16(false, a, false, b0, (short)0, acc0, false, false);
        acc1 = __builtin_amdgcn_wmma_f32_16x16x32_bf16(false, a, false, b1, (short)0, acc1, false, false);
        acc2 = __builtin_amdgcn_wmma_f32_16x16x32_bf16(false, a, false, b2, (short)0, acc2, false, false);
        acc3 = __builtin_amdgcn_wmma_f32_16x16x32_bf16(false, a, false, b3, (short)0, acc3, false, false);
    }
    const int lane = threadIdx.x & 31;
    const int n = lane & 15;
    v8f accs[4] = {acc0, acc1, acc2, acc3};
#pragma unroll
    for (int j = 0; j < 4; ++j) {
#pragma unroll
        for (int r = 0; r < 8; ++r) {
            int m = m0 + r + ((lane >> 4) << 3);
            xw[(size_t)m * HID + n0 + j * 16 + n] = accs[j][r];
        }
    }
}

// ---------------------------------------------------------------------------
// One recurrence step: h_new = tanh(xw_t + h @ Wh + bh)   M=L*B=256, N=K=1536
// 16x64 per wave (A reused 4x per chunk, 4 WMMA chains); prefetch next B rows.
// Writes f32 state (final output) + bf16 copy (next step's A operand).
// ---------------------------------------------------------------------------
__global__ void rnn_step_kernel(const __bf16* __restrict__ h_in_bf,
                                const __bf16* __restrict__ WhT,
                                const float* __restrict__ xw_t,
                                const float* __restrict__ bh,
                                float* __restrict__ h_out_f,
                                __bf16* __restrict__ h_out_bf) {
    const int NG = HID / 64;  // 24
    const int gw = (int)((blockIdx.x * blockDim.x + threadIdx.x) >> 5);
    const int m0 = (gw / NG) * 16;
    const int n0 = (gw % NG) * 64;
    const int lane = threadIdx.x & 31;

    v8f acc0 = {}, acc1 = {}, acc2 = {}, acc3 = {};
    for (int k0 = 0; k0 < HID; k0 += 32) {
        // speculative prefetch of next K-chunk of B (-> global_prefetch_b8)
        __builtin_prefetch(WhT + (size_t)(n0 + (lane & 15)) * HID + k0 + 32, 0, 1);
        v16bf a  = load_a_frag(h_in_bf, m0, k0, HID);
        v16bf b0 = load_b_frag(WhT, n0,      k0, HID);
        v16bf b1 = load_b_frag(WhT, n0 + 16, k0, HID);
        v16bf b2 = load_b_frag(WhT, n0 + 32, k0, HID);
        v16bf b3 = load_b_frag(WhT, n0 + 48, k0, HID);
        acc0 = __builtin_amdgcn_wmma_f32_16x16x32_bf16(false, a, false, b0, (short)0, acc0, false, false);
        acc1 = __builtin_amdgcn_wmma_f32_16x16x32_bf16(false, a, false, b1, (short)0, acc1, false, false);
        acc2 = __builtin_amdgcn_wmma_f32_16x16x32_bf16(false, a, false, b2, (short)0, acc2, false, false);
        acc3 = __builtin_amdgcn_wmma_f32_16x16x32_bf16(false, a, false, b3, (short)0, acc3, false, false);
    }
    v8f accs[4] = {acc0, acc1, acc2, acc3};
#pragma unroll
    for (int j = 0; j < 4; ++j) {
        const int n = n0 + j * 16 + (lane & 15);
        const float bias = bh[n];
#pragma unroll
        for (int r = 0; r < 8; ++r) {
            int row  = m0 + r + ((lane >> 4) << 3);   // row = l*B + b
            int bidx = row & (BATCH - 1);             // xw_t broadcast over layers
            float v = tanhf(accs[j][r] + xw_t[(size_t)bidx * HID + n] + bias);
            h_out_f[(size_t)row * HID + n]  = v;
            h_out_bf[(size_t)row * HID + n] = (__bf16)v;
        }
    }
}

// ---------------------------------------------------------------------------
// out_const = hidden0[-1] @ Wy + by   M=128 N=10000 K=1536  (source-bug path)
// 16x80 per wave (625 N-tiles = 5 * 125 groups)
// ---------------------------------------------------------------------------
__global__ void logits_gemm_kernel(const __bf16* __restrict__ h0bf,
                                   const __bf16* __restrict__ WyT,
                                   const float* __restrict__ by,
                                   float* __restrict__ outc) {
    const int NG = VOCAB / 80;  // 125 groups of 5 tiles
    const int gw = (int)((blockIdx.x * blockDim.x + threadIdx.x) >> 5);
    const int m0 = (gw / NG) * 16;
    const int n0 = (gw % NG) * 80;

    v8f acc[5] = {};
    for (int k0 = 0; k0 < HID; k0 += 32) {
        v16bf a = load_a_frag(h0bf, m0, k0, HID);
#pragma unroll
        for (int j = 0; j < 5; ++j) {
            v16bf b = load_b_frag(WyT, n0 + j * 16, k0, HID);
            acc[j] = __builtin_amdgcn_wmma_f32_16x16x32_bf16(false, a, false, b,
                                                             (short)0, acc[j], false, false);
        }
    }
    const int lane = threadIdx.x & 31;
#pragma unroll
    for (int j = 0; j < 5; ++j) {
        const int n = n0 + j * 16 + (lane & 15);
        const float bias = by[n];
#pragma unroll
        for (int r = 0; r < 8; ++r) {
            int m = m0 + r + ((lane >> 4) << 3);
            outc[(size_t)m * VOCAB + n] = acc[j][r] + bias;
        }
    }
}

// ---------------------------------------------------------------------------
// logits[t] = out_const for all t (655 MB store, float4 vectorized)
// grid.x * 256 == B*V/4 exactly; grid.y == T
// ---------------------------------------------------------------------------
__global__ void broadcast_logits_kernel(const float* __restrict__ outc,
                                        float* __restrict__ out) {
    const size_t bv4 = (size_t)BATCH * VOCAB / 4;
    size_t i = (size_t)blockIdx.x * blockDim.x + threadIdx.x;  // float4 idx
    size_t t = blockIdx.y;
    v4f v = ((const v4f*)outc)[i];
    ((v4f*)out)[t * bv4 + i] = v;
}

// ---------------------------------------------------------------------------
extern "C" void kernel_launch(void* const* d_in, const int* in_sizes, int n_in,
                              void* d_out, int out_size, void* d_ws, size_t ws_size,
                              hipStream_t stream) {
    const float* lut = (const float*)d_in[0];
    const float* Wx  = (const float*)d_in[1];
    const float* Wh  = (const float*)d_in[2];
    const float* Wy  = (const float*)d_in[3];
    const float* bh  = (const float*)d_in[4];
    const float* by  = (const float*)d_in[5];
    const float* h0  = (const float*)d_in[6];
    const int*   inp = (const int*)d_in[7];
    float* out = (float*)d_out;

    // workspace carve-out (aligned 256B)
    char* p = (char*)d_ws;
    auto carve = [&](size_t bytes) {
        char* r = p;
        p += (bytes + 255) & ~(size_t)255;
        return r;
    };
    __bf16* WxT  = (__bf16*)carve((size_t)HID * EMB * 2);        // N x K
    __bf16* WhT  = (__bf16*)carve((size_t)HID * HID * 2);
    __bf16* WyT  = (__bf16*)carve((size_t)VOCAB * HID * 2);
    float*  xw   = (float*)carve((size_t)T_STEPS * BATCH * HID * 4);
    __bf16* hbfA = (__bf16*)carve((size_t)LAYERS * BATCH * HID * 2);
    __bf16* hbfB = (__bf16*)carve((size_t)LAYERS * BATCH * HID * 2);
    float*  hf   = (float*)carve((size_t)LAYERS * BATCH * HID * 4);
    __bf16* h0bf = (__bf16*)carve((size_t)BATCH * HID * 2);
    float*  outc = (float*)carve((size_t)BATCH * VOCAB * 4);

    // 1) weights -> bf16, transposed
    cvt_transpose_kernel<<<1024, 256, 0, stream>>>(Wx, WxT, EMB, HID);
    cvt_transpose_kernel<<<1024, 256, 0, stream>>>(Wh, WhT, HID, HID);
    cvt_transpose_kernel<<<4096, 256, 0, stream>>>(Wy, WyT, HID, VOCAB);

    // 2) xw = emb @ Wx   (1024 M-tiles * 24 N-groups = 24576 waves)
    embed_xw_gemm_kernel<<<3072, 256, 0, stream>>>(lut, inp, WxT, xw);

    // 3) hidden init (f32 state + bf16 copy) and bf16 of hidden[-1] for logits
    init_hidden_kernel<<<512, 256, 0, stream>>>(h0, hf, hbfA,
                                                (size_t)LAYERS * BATCH * HID);
    cvt_bf16_kernel<<<256, 256, 0, stream>>>(h0 + (size_t)(LAYERS - 1) * BATCH * HID,
                                             h0bf, (size_t)BATCH * HID);

    // 4) sequential scan: 128 dependent steps (16 M-tiles * 24 N-groups = 384 waves)
    __bf16* bin = hbfA;
    __bf16* bout = hbfB;
    for (int t = 0; t < T_STEPS; ++t) {
        rnn_step_kernel<<<48, 256, 0, stream>>>(bin, WhT,
                                                xw + (size_t)t * BATCH * HID,
                                                bh, hf, bout);
        __bf16* tmp = bin; bin = bout; bout = tmp;
    }

    // 5) h_final -> second output region
    copy_f32_kernel<<<512, 256, 0, stream>>>(hf, out + (size_t)T_STEPS * BATCH * VOCAB,
                                             (size_t)LAYERS * BATCH * HID);

    // 6) out_const = hidden0[-1] @ Wy + by  (8 M-tiles * 125 N-groups = 1000 waves)
    logits_gemm_kernel<<<125, 256, 0, stream>>>(h0bf, WyT, by, outc);

    // 7) broadcast to all T timesteps (the bandwidth floor: 655 MB of stores)
    dim3 gbc(1250, T_STEPS);
    broadcast_logits_kernel<<<gbc, 256, 0, stream>>>(outc, out);
}
